// LatentGPT_79293686219031
// MI455X (gfx1250) — matrix operand order repaired and checked by reading, not compiled, not run on hardware
//
#include <hip/hip_runtime.h>
#include <hip/hip_bf16.h>
#include <math.h>

// ---------------- model dims ----------------
#define BB 4
#define TT 512
#define SS 77
#define DD 1024
#define HH 16
#define FF 4096
#define VV 16384
#define CDM 512
#define LL 8
#define DH 64
#define EPS 1e-5f

typedef __bf16 bf16_t;
typedef __attribute__((ext_vector_type(2)))  __bf16 v2bf;
typedef __attribute__((ext_vector_type(16))) __bf16 v16bf;
typedef __attribute__((ext_vector_type(8)))  float  v8f;

// pack two f32 into one dword of bf16 (lo, hi) -> v_cvt_pk_bf16_f32
__device__ __forceinline__ unsigned pack_bf2(float lo, float hi) {
    v2bf v;
    v[0] = (bf16_t)lo;
    v[1] = (bf16_t)hi;
    return __builtin_bit_cast(unsigned, v);
}

struct Frag32B { uint4 lo, hi; };   // 32 bytes -> v16bf

// =====================================================================
// GEMM:  C[M,N] = epilogue( A[M,K] @ W[K,N] + bias )
//   op = 0: +bias (bias may be nullptr) | 1: exact GELU(+bias) | 2: +bias+res
// Block 256 thr (8 waves). Tile BM=128,BN=128,BK=32; wave grid 2x4,
// each wave 64x32 = 4x2 WMMA tiles, f32 accum. Double-buffered LDS,
// pipeline loop unrolled x2 (K/32 is even for all call sites) so the
// buffer index is a literal and LDS offsets are immediates.
// A tile K-major packed (2 bf16/dword); W tile stored TRANSPOSED
// ([col][k] packed) so both A and B fragments load as 2x ds_load_b128.
// Requires K%64==0, N%128==0; M guarded.
// =====================================================================
__global__ __launch_bounds__(256)
void gemm_bf16_wmma(const float* __restrict__ A, const float* __restrict__ W,
                    const float* __restrict__ bias, const float* __restrict__ res,
                    float* __restrict__ C, int M, int N, int K, int op)
{
    // [buf][row][kpair]  pitch 20 dwords (80B) keeps 16B alignment + pad
    __shared__ unsigned As32[2][128][20];
    __shared__ unsigned Bst32[2][128][20];

    const int tid  = threadIdx.x;
    const int lane = tid & 31;
    const int wave = tid >> 5;
    const int wm   = wave >> 2;      // 0..1 -> 64 rows
    const int wn   = wave & 3;       // 0..3 -> 32 cols
    const int m0   = blockIdx.y * 128;
    const int n0   = blockIdx.x * 128;
    const int l16  = lane & 15;
    const int hsel = lane >> 4;

    // ---- staging lambdas -------------------------------------------
    auto stageA = [&](int buf, int k0) {
        const int row  = tid >> 1;           // 0..127
        const int cseg = (tid & 1) * 16;     // 0 / 16 (floats)
        const int gr   = m0 + row;
        float4 f0, f1, f2, f3;
        if (gr < M) {
            const float4* src = (const float4*)(A + (size_t)gr * K + k0 + cseg);
            f0 = src[0]; f1 = src[1]; f2 = src[2]; f3 = src[3];
        } else {
            f0 = f1 = f2 = f3 = make_float4(0.f, 0.f, 0.f, 0.f);
        }
        uint4 p0, p1;
        p0.x = pack_bf2(f0.x, f0.y); p0.y = pack_bf2(f0.z, f0.w);
        p0.z = pack_bf2(f1.x, f1.y); p0.w = pack_bf2(f1.z, f1.w);
        p1.x = pack_bf2(f2.x, f2.y); p1.y = pack_bf2(f2.z, f2.w);
        p1.z = pack_bf2(f3.x, f3.y); p1.w = pack_bf2(f3.z, f3.w);
        uint4* dst = (uint4*)&As32[buf][row][cseg >> 1];  // cseg/2 dwords
        dst[0] = p0; dst[1] = p1;
    };
    auto stageB = [&](int buf, int k0) {
        const int kp   = tid >> 4;           // 0..15 : K pair (rows 2kp,2kp+1)
        const int nseg = (tid & 15) * 8;     // 8 cols per thread
        const float4* r0 = (const float4*)(W + (size_t)(k0 + 2 * kp) * N + n0 + nseg);
        const float4* r1 = (const float4*)(W + (size_t)(k0 + 2 * kp + 1) * N + n0 + nseg);
        float4 a0 = r0[0], a1 = r0[1];
        float4 b0 = r1[0], b1 = r1[1];
        // transposed scatter: Bst[col][kp] = (lo=row k, hi=row k+1)
        Bst32[buf][nseg + 0][kp] = pack_bf2(a0.x, b0.x);
        Bst32[buf][nseg + 1][kp] = pack_bf2(a0.y, b0.y);
        Bst32[buf][nseg + 2][kp] = pack_bf2(a0.z, b0.z);
        Bst32[buf][nseg + 3][kp] = pack_bf2(a0.w, b0.w);
        Bst32[buf][nseg + 4][kp] = pack_bf2(a1.x, b1.x);
        Bst32[buf][nseg + 5][kp] = pack_bf2(a1.y, b1.y);
        Bst32[buf][nseg + 6][kp] = pack_bf2(a1.z, b1.z);
        Bst32[buf][nseg + 7][kp] = pack_bf2(a1.w, b1.w);
    };

    v8f acc[4][2];
    #pragma unroll
    for (int i = 0; i < 4; ++i)
        #pragma unroll
        for (int j = 0; j < 2; ++j)
            acc[i][j] = (v8f){};

    const int nk = K >> 5;                 // even at every call site

    stageA(0, 0);
    stageB(0, 0);
    __syncthreads();

    // one pipeline stage: frags(buf) -> stage next(1-buf) -> 8 WMMAs -> barrier
    auto iter = [&](int buf, int kk) {
        v16bf afr[4];
        #pragma unroll
        for (int i = 0; i < 4; ++i) {
            const int r = wm * 64 + i * 16 + l16;
            const uint4* p = (const uint4*)&As32[buf][r][0];
            Frag32B fr;
            fr.lo = p[hsel];          // dwords hsel*4..+3  : K = hsel*8..+7
            fr.hi = p[2 + hsel];      // dwords 8+hsel*4..  : K = 16+hsel*8..+7
            afr[i] = __builtin_bit_cast(v16bf, fr);
        }
        v16bf bfr[2];
        #pragma unroll
        for (int j = 0; j < 2; ++j) {
            const int c = wn * 32 + j * 16 + l16;
            const uint4* p = (const uint4*)&Bst32[buf][c][0];
            Frag32B fr;
            fr.lo = p[hsel];          // dwords hsel*4..+3  : K = hsel*8..+7
            fr.hi = p[2 + hsel];      // dwords 8+hsel*4..  : K = 16+hsel*8..+7
            bfr[j] = __builtin_bit_cast(v16bf, fr);
        }

        if (kk + 1 < nk) {
            if (kk + 2 < nk) {
                const int kf = (kk + 2) * 32;
                const int gr = m0 + (tid >> 1);
                if (gr < M)
                    __builtin_prefetch(A + (size_t)gr * K + kf, 0, 1);
                __builtin_prefetch(W + (size_t)(kf + (tid >> 4) * 2) * N + n0, 0, 1);
            }
            stageA(1 - buf, (kk + 1) * 32);
            stageB(1 - buf, (kk + 1) * 32);
        }

        #pragma unroll
        for (int i = 0; i < 4; ++i)
            #pragma unroll
            for (int j = 0; j < 2; ++j)
                acc[i][j] = __builtin_amdgcn_wmma_f32_16x16x32_bf16(
                    false, afr[i], false, bfr[j], (short)0, acc[i][j], false, false);

        __syncthreads();
    };

    for (int kk = 0; kk < nk; kk += 2) {
        iter(0, kk);          // buffer index is a literal -> const LDS offsets
        iter(1, kk + 1);
    }

    // ---- epilogue + store (C/D: VGPR r -> M=r / M=8+r) -------------
    #pragma unroll
    for (int i = 0; i < 4; ++i) {
        #pragma unroll
        for (int j = 0; j < 2; ++j) {
            const int col     = n0 + wn * 32 + j * 16 + l16;
            const int rowBase = m0 + wm * 64 + i * 16 + hsel * 8;
            const float bv_ = bias ? bias[col] : 0.0f;
            #pragma unroll
            for (int r = 0; r < 8; ++r) {
                const int row = rowBase + r;
                if (row < M) {
                    float v = acc[i][j][r] + bv_;
                    if (op == 1)      v = 0.5f * v * (1.0f + erff(v * 0.70710678118f));
                    else if (op == 2) v += res[(size_t)row * N + col];
                    C[(size_t)row * N + col] = v;
                }
            }
        }
    }
}

// =====================================================================
// Embedding + sinusoidal PE:  x[b,t,:] = tok_emb[tok] + pe(t)
// =====================================================================
__global__ __launch_bounds__(256)
void embed_kernel(const int* __restrict__ tokens, const float* __restrict__ emb,
                  float* __restrict__ x)
{
    const int bt  = blockIdx.x;          // 0 .. B*T-1
    const int t   = bt % TT;
    const int tok = tokens[bt];
    const float tf = (float)t;
    #pragma unroll
    for (int i = 0; i < 4; ++i) {
        const int d    = threadIdx.x + i * 256;
        const int pair = d >> 1;
        const float ang = tf * __expf((float)(2 * pair) * (-9.210340371976184f / (float)DD));
        const float pe  = (d & 1) ? __cosf(ang) : __sinf(ang);
        x[(size_t)bt * DD + d] = emb[(size_t)tok * DD + d] + pe;
    }
}

// =====================================================================
// LayerNorm over last dim (D=1024), population variance, eps=1e-5
// =====================================================================
__global__ __launch_bounds__(256)
void ln_kernel(const float* __restrict__ x, const float* __restrict__ g,
               const float* __restrict__ b, float* __restrict__ y)
{
    __shared__ float s1[256], s2[256];
    const int row = blockIdx.x;
    const int tid = threadIdx.x;
    const float* xr = x + (size_t)row * DD;

    float sum = 0.f, sq = 0.f;
    #pragma unroll
    for (int i = 0; i < 4; ++i) {
        const float v = xr[tid + i * 256];
        sum += v; sq += v * v;
    }
    s1[tid] = sum; s2[tid] = sq;
    __syncthreads();
    for (int off = 128; off > 0; off >>= 1) {
        if (tid < off) { s1[tid] += s1[tid + off]; s2[tid] += s2[tid + off]; }
        __syncthreads();
    }
    const float mean = s1[0] * (1.0f / DD);
    const float var  = s2[0] * (1.0f / DD) - mean * mean;
    const float rstd = rsqrtf(var + EPS);
    #pragma unroll
    for (int i = 0; i < 4; ++i) {
        const int d = tid + i * 256;
        y[(size_t)row * DD + d] = (xr[d] - mean) * rstd * g[d] + b[d];
    }
}

// =====================================================================
// y = a + b (elementwise)
// =====================================================================
__global__ __launch_bounds__(256)
void add_kernel(const float* __restrict__ a, const float* __restrict__ b,
                float* __restrict__ y, size_t n)
{
    for (size_t i = (size_t)blockIdx.x * blockDim.x + threadIdx.x; i < n;
         i += (size_t)gridDim.x * blockDim.x)
        y[i] = a[i] + b[i];
}

// =====================================================================
// Attention (one block per (b, h, t) query row).
//   q: [B, qlen, D], k/v: [B, kvlen, D] (head h at cols h*DH..),
//   o: [B, qlen, D].  causal: keys restricted to s <= t.
// Block = 128 threads. kvlen <= 512.
// =====================================================================
__global__ __launch_bounds__(128)
void attn_kernel(const float* __restrict__ q, const float* __restrict__ k,
                 const float* __restrict__ v, float* __restrict__ o,
                 int qlen, int kvlen, int causal)
{
    __shared__ float qs[DH];
    __shared__ float ls[512];
    __shared__ float red[128];

    const int blk = blockIdx.x;
    const int t   = blk % qlen;
    const int h   = (blk / qlen) % HH;
    const int b   = blk / (qlen * HH);
    const int tid = threadIdx.x;

    const float* qrow = q + ((size_t)(b * qlen + t) * DD) + h * DH;
    if (tid < DH) qs[tid] = qrow[tid];
    __syncthreads();

    const int nkey = causal ? (t + 1) : kvlen;
    const float scale = 0.125f;   // 1/sqrt(64)

    for (int s = tid; s < nkey; s += 128) {
        const float* krow = k + ((size_t)(b * kvlen + s) * DD) + h * DH;
        float dot = 0.f;
        #pragma unroll 8
        for (int d = 0; d < DH; ++d) dot += qs[d] * krow[d];
        ls[s] = dot * scale;
    }
    __syncthreads();

    // max
    float m = -1e30f;
    for (int s = tid; s < nkey; s += 128) m = fmaxf(m, ls[s]);
    red[tid] = m;
    __syncthreads();
    for (int off = 64; off > 0; off >>= 1) {
        if (tid < off) red[tid] = fmaxf(red[tid], red[tid + off]);
        __syncthreads();
    }
    m = red[0];
    __syncthreads();

    // exp + sum
    float sm = 0.f;
    for (int s = tid; s < nkey; s += 128) {
        const float e = __expf(ls[s] - m);
        ls[s] = e; sm += e;
    }
    red[tid] = sm;
    __syncthreads();
    for (int off = 64; off > 0; off >>= 1) {
        if (tid < off) red[tid] += red[tid + off];
        __syncthreads();
    }
    const float inv = 1.0f / red[0];
    __syncthreads();

    // out[d] = inv * sum_s ls[s] * v[s][d]   (2 threads per d)
    const int d    = tid & (DH - 1);
    const int half = tid >> 6;
    float accv = 0.f;
    for (int s = half; s < nkey; s += 2)
        accv += ls[s] * v[((size_t)(b * kvlen + s) * DD) + h * DH + d];
    red[tid] = accv;
    __syncthreads();
    if (half == 0)
        o[((size_t)(b * qlen + t) * DD) + h * DH + d] = (red[tid] + red[tid + 64]) * inv;
}

// =====================================================================
// Host orchestration
// =====================================================================
static inline void run_gemm(const float* A, const float* W, const float* bias,
                            const float* res, float* C, int M, int N, int K,
                            int op, hipStream_t s)
{
    dim3 grid(N / 128, (M + 127) / 128);
    gemm_bf16_wmma<<<grid, 256, 0, s>>>(A, W, bias, res, C, M, N, K, op);
}

extern "C" void kernel_launch(void* const* d_in, const int* in_sizes, int n_in,
                              void* d_out, int out_size, void* d_ws, size_t ws_size,
                              hipStream_t stream)
{
    const int*   tokens = (const int*)  d_in[0];
    const float* ctx    = (const float*)d_in[1];   // [B,S,CD]
    const float* tokemb = (const float*)d_in[2];
    const float* Wq  = (const float*)d_in[3];  const float* bq  = (const float*)d_in[4];
    const float* Wk  = (const float*)d_in[5];  const float* bk  = (const float*)d_in[6];
    const float* Wv  = (const float*)d_in[7];  const float* bv  = (const float*)d_in[8];
    const float* Wo  = (const float*)d_in[9];  const float* bo  = (const float*)d_in[10];
    const float* cWq = (const float*)d_in[11]; const float* cbq = (const float*)d_in[12];
    const float* cWk = (const float*)d_in[13]; const float* cbk = (const float*)d_in[14];
    const float* cWv = (const float*)d_in[15]; const float* cbv = (const float*)d_in[16];
    const float* cWo = (const float*)d_in[17]; const float* cbo = (const float*)d_in[18];
    const float* clng= (const float*)d_in[19]; const float* clnb= (const float*)d_in[20];
    const float* ln1g= (const float*)d_in[21]; const float* ln1b= (const float*)d_in[22];
    const float* ln2g= (const float*)d_in[23]; const float* ln2b= (const float*)d_in[24];
    const float* W1  = (const float*)d_in[25]; const float* b1  = (const float*)d_in[26];
    const float* W2  = (const float*)d_in[27]; const float* b2  = (const float*)d_in[28];
    const float* lnog= (const float*)d_in[29]; const float* lnob= (const float*)d_in[30];
    const float* Whd = (const float*)d_in[31];

    const int MT = BB * TT;        // 2048 token rows
    const int MS = BB * SS;        // 308 context rows

    // workspace carve-up (f32)
    size_t off = 0;
    float* wsf = (float*)d_ws;
    auto carve = [&](size_t n) { float* p = wsf + off; off += n; return p; };
    float* x    = carve((size_t)MT * DD);
    float* h    = carve((size_t)MT * DD);
    float* qb   = carve((size_t)MT * DD);
    float* kb2  = carve((size_t)MT * DD);
    float* vb   = carve((size_t)MT * DD);
    float* attn = carve((size_t)MT * DD);
    float* outb = carve((size_t)MT * DD);
    float* t1   = carve((size_t)MT * DD);
    float* xn   = carve((size_t)MT * DD);
    float* cq   = carve((size_t)MT * DD);
    float* co   = carve((size_t)MT * DD);
    float* h2   = carve((size_t)MT * DD);
    float* ck   = carve((size_t)MS * DD);
    float* cv   = carve((size_t)MS * DD);
    float* mid  = carve((size_t)MT * FF);
    (void)ws_size; (void)n_in; (void)in_sizes; (void)out_size;

    // x = tok_emb[tokens] + PE
    embed_kernel<<<MT, 256, 0, stream>>>(tokens, tokemb, x);

    for (int l = 0; l < LL; ++l) {
        const size_t wdd = (size_t)l * DD * DD;
        const size_t wcd = (size_t)l * CDM * DD;
        const size_t wdf = (size_t)l * DD * FF;
        const size_t bd  = (size_t)l * DD;
        const size_t bf  = (size_t)l * FF;

        // h = ln1(x)
        ln_kernel<<<MT, 256, 0, stream>>>(x, ln1g + bd, ln1b + bd, h);
        // QKV
        run_gemm(h, Wq + wdd, bq + bd, nullptr, qb,  MT, DD, DD, 0, stream);
        run_gemm(h, Wk + wdd, bk + bd, nullptr, kb2, MT, DD, DD, 0, stream);
        run_gemm(h, Wv + wdd, bv + bd, nullptr, vb,  MT, DD, DD, 0, stream);
        // causal self-attention
        attn_kernel<<<BB * HH * TT, 128, 0, stream>>>(qb, kb2, vb, attn, TT, TT, 1);
        // out = attn @ Wo + bo
        run_gemm(attn, Wo + wdd, bo + bd, nullptr, outb, MT, DD, DD, 0, stream);
        // t1 = h + out ; xn = cross_ln(t1)
        add_kernel<<<1024, 256, 0, stream>>>(h, outb, t1, (size_t)MT * DD);
        ln_kernel<<<MT, 256, 0, stream>>>(t1, clng + bd, clnb + bd, xn);
        // cross QKV
        run_gemm(xn,  cWq + wdd, cbq + bd, nullptr, cq, MT, DD, DD,  0, stream);
        run_gemm(ctx, cWk + wcd, cbk + bd, nullptr, ck, MS, DD, CDM, 0, stream);
        run_gemm(ctx, cWv + wcd, cbv + bd, nullptr, cv, MS, DD, CDM, 0, stream);
        // cross attention
        attn_kernel<<<BB * HH * TT, 128, 0, stream>>>(cq, ck, cv, co, TT, SS, 0);
        // out = out + co @ cWo + cbo   (residual epilogue, in-place safe)
        run_gemm(co, cWo + wdd, cbo + bd, outb, outb, MT, DD, DD, 2, stream);
        // x = x + out
        add_kernel<<<1024, 256, 0, stream>>>(x, outb, x, (size_t)MT * DD);
        // FFN: h2 = ln2(x); mid = gelu(h2@W1+b1); x = x + mid@W2+b2
        ln_kernel<<<MT, 256, 0, stream>>>(x, ln2g + bd, ln2b + bd, h2);
        run_gemm(h2,  W1 + wdf, b1 + bf, nullptr, mid, MT, FF, DD, 1, stream);
        run_gemm(mid, W2 + wdf, b2 + bd, x,       x,   MT, DD, FF, 2, stream);
    }

    // logits = ln(x) @ Whead
    ln_kernel<<<MT, 256, 0, stream>>>(x, lnog, lnob, h);
    run_gemm(h, Whd, nullptr, nullptr, (float*)d_out, MT, VV, DD, 0, stream);
}
